// ClusterMemory_double_27814208209142
// MI455X (gfx1250) — matrix operand order repaired and checked by reading, not compiled, not run on hardware
//
#include <hip/hip_runtime.h>
#include <hip/hip_bf16.h>

typedef __attribute__((ext_vector_type(16))) _Float16 v16h;
typedef __attribute__((ext_vector_type(8)))  float    v8f;

#define BATCH   256
#define NPROTO  8192
#define DIM     768
#define INV_T   14.285714285714286f   /* 1/0.07 */
#define SHIFT   14.285714285714286f   /* dot<=1 -> logit<=SHIFT -> exp()<=1 */
#define MBLK    32
#define NCHUNK  1024
#define LDSP    776                   /* 768 + 8 halfs pad: avoids 16-way LDS bank conflict */

// workspace layout (bytes)
#define WS_SLOTS   0                  /* 3 * 65536 floats = 786432 B          */
#define WS_FEATH   786432             /* 2*256*768 halfs  = 786432 B          */
#define WS_MEMH    1572864            /* 2*8192*768 halfs = 25165824 B        */
#define WS_NEED    (1572864 + 25165824)

union AFrag { v16h h; uint4 q[2]; };
union H8    { _Float16 h[8]; uint4 q; };

// ---------------------------------------------------------------------------
// Kernel 1: L2-normalize anchors, store as f16 into workspace.
// grid = 512 (2 modalities x 256 rows), block = 256
// ---------------------------------------------------------------------------
__global__ void supcon_prep(const float* __restrict__ rgb,
                            const float* __restrict__ ir,
                            _Float16* __restrict__ featH) {
  const int row = blockIdx.x & 255;
  const int mod = blockIdx.x >> 8;
  const float* src = (mod ? ir : rgb) + (size_t)row * DIM;

  float ss = 0.0f;
  for (int c = threadIdx.x; c < DIM; c += 256) {
    float x = src[c];
    ss += x * x;
  }
  #pragma unroll
  for (int m = 16; m >= 1; m >>= 1) ss += __shfl_xor(ss, m, 32);
  __shared__ float red[8];
  if ((threadIdx.x & 31) == 0) red[threadIdx.x >> 5] = ss;
  __syncthreads();
  float tot = red[0] + red[1] + red[2] + red[3] + red[4] + red[5] + red[6] + red[7];
  float inv = 1.0f / fmaxf(sqrtf(tot), 1e-12f);

  _Float16* dst = featH + (size_t)(mod * BATCH + row) * DIM;
  for (int c = threadIdx.x; c < DIM; c += 256)
    dst[c] = (_Float16)(src[c] * inv);
}

// ---------------------------------------------------------------------------
// Kernel 1b: bulk-convert the two prototype banks f32 -> f16 (already unit
// norm). One pass: 50 MB read + 25 MB write. 8 elems/thread, vectorized.
// grid = (3072, 2), block = 256
// ---------------------------------------------------------------------------
__global__ void supcon_convbank(const float* __restrict__ bankA,
                                const float* __restrict__ bankB,
                                _Float16* __restrict__ memH) {
  const int bank = blockIdx.y;
  const float* src = (bank ? bankB : bankA);
  _Float16* dst = memH + (size_t)bank * NPROTO * DIM;
  size_t off = ((size_t)blockIdx.x * 256 + threadIdx.x) * 8;
  float4 f0 = *(const float4*)(src + off);
  float4 f1 = *(const float4*)(src + off + 4);
  H8 o;
  o.h[0]=(_Float16)f0.x; o.h[1]=(_Float16)f0.y; o.h[2]=(_Float16)f0.z; o.h[3]=(_Float16)f0.w;
  o.h[4]=(_Float16)f1.x; o.h[5]=(_Float16)f1.y; o.h[6]=(_Float16)f1.z; o.h[7]=(_Float16)f1.w;
  *(uint4*)(dst + off) = o.q;
}

// ---------------------------------------------------------------------------
// Kernel 2: fused GEMM (WMMA f16 -> f32) + shifted-exp / positive stats.
// grid = (8 n-chunks, 8 m-blocks, 4 pairings), block = 256 (8 waves)
// Each wave writes its own partial slot -> fully deterministic (no atomics).
// slots layout (floats): [0 .. 64K) sumexp, [64K .. 128K) possum, [128K..192K) cnt
// slot index = (p*256 + anchor)*64 + (nchunk*8 + wave)
// PRECONV=true : B fragments read pre-converted f16 bank (no cvts in loop)
// PRECONV=false: B fragments read f32 bank, convert in registers
// ---------------------------------------------------------------------------
template <bool PRECONV>
__global__ __launch_bounds__(256, 1)
void supcon_gemm(const _Float16* __restrict__ featH,
                 const _Float16* __restrict__ memH,   // f16 banks (PRECONV)
                 const float* __restrict__ memA,      // features_rgb (f32)
                 const float* __restrict__ memB,      // features_ir  (f32)
                 const int* __restrict__ tgt_rgb, const int* __restrict__ tgt_ir,
                 const int* __restrict__ pl_rgb,  const int* __restrict__ pl_ir,
                 float* __restrict__ slots) {
  const int nchunk = blockIdx.x;        // 0..7
  const int mb     = blockIdx.y;        // 0..7
  const int p      = blockIdx.z;        // 0..3
  const int modF   = (p == 0 || p == 2) ? 0 : 1;            // rgb feats for p0,p2
  const int bankI  = (p == 0 || p == 3) ? 0 : 1;            // rgb bank for p0,p3
  const float*    memF = bankI ? memB : memA;
  const _Float16* memh = memH + (size_t)bankI * NPROTO * DIM;
  const int* tgts  = (p == 0 || p == 2) ? tgt_rgb : tgt_ir;
  const int* mlab  = (p == 0 || p == 3) ? pl_rgb  : pl_ir;

  __shared__ _Float16 As[MBLK * LDSP];  // 49,664 B

  // cooperative A-tile stage: 32 rows x 768 halfs, vectorized as uint4
  {
    const _Float16* fr = featH + (size_t)(modF * BATCH + mb * MBLK) * DIM;
    for (int j = threadIdx.x; j < MBLK * (DIM / 8); j += 256) {
      int r = j / (DIM / 8), c = j % (DIM / 8);
      *(uint4*)(&As[r * LDSP + c * 8]) = *(const uint4*)(fr + (size_t)r * DIM + c * 8);
    }
  }
  __syncthreads();

  const int lane  = threadIdx.x & 31;
  const int wave  = threadIdx.x >> 5;
  const int l15   = lane & 15;
  const int khalf = lane >> 4;          // 0 or 1 (WMMA K-half per lane group)

  // register-cache anchor labels for my 16 C-rows (matches C VGPR layout)
  int lab[16];
  #pragma unroll
  for (int s = 0; s < 2; ++s)
    #pragma unroll
    for (int v = 0; v < 8; ++v)
      lab[s * 8 + v] = tgts[mb * MBLK + s * 16 + khalf * 8 + v];

  float eacc[16], pacc[16], cacc[16];
  #pragma unroll
  for (int i = 0; i < 16; ++i) { eacc[i] = 0.f; pacc[i] = 0.f; cacc[i] = 0.f; }

  const int ncBase = nchunk * NCHUNK + wave * 128;

  for (int t = 0; t < 8; ++t) {
    const int myn = ncBase + t * 16 + l15;
    const int ml  = mlab[myn];

    v8f c0 = {}; v8f c1 = {};

    if (PRECONV) {
      const _Float16* rowB = memh + (size_t)myn * DIM + khalf * 8;
      if (myn + 16 < NPROTO)
        __builtin_prefetch(memh + (size_t)(myn + 16) * DIM + khalf * 16, 0, 0);
      #pragma unroll 4
      for (int k0 = 0; k0 < DIM; k0 += 32) {
        AFrag b;
        b.q[0] = *(const uint4*)(rowB + k0);
        b.q[1] = *(const uint4*)(rowB + k0 + 16);
        AFrag a0, a1;
        const _Float16* A0 = &As[l15 * LDSP + k0 + khalf * 8];
        a0.q[0] = *(const uint4*)(A0);
        a0.q[1] = *(const uint4*)(A0 + 16);
        const _Float16* A1 = &As[(16 + l15) * LDSP + k0 + khalf * 8];
        a1.q[0] = *(const uint4*)(A1);
        a1.q[1] = *(const uint4*)(A1 + 16);
        c0 = __builtin_amdgcn_wmma_f32_16x16x32_f16(false, a0.h, false, b.h, (short)0, c0, false, false);
        c1 = __builtin_amdgcn_wmma_f32_16x16x32_f16(false, a1.h, false, b.h, (short)0, c1, false, false);
      }
    } else {
      const float* rowB = memF + (size_t)myn * DIM + khalf * 8;
      if (myn + 16 < NPROTO)
        __builtin_prefetch(memF + (size_t)(myn + 16) * DIM + khalf * 16, 0, 0);
      #pragma unroll 4
      for (int k0 = 0; k0 < DIM; k0 += 32) {
        float4 f0 = *(const float4*)(rowB + k0);
        float4 f1 = *(const float4*)(rowB + k0 + 4);
        float4 f2 = *(const float4*)(rowB + k0 + 16);
        float4 f3 = *(const float4*)(rowB + k0 + 20);
        v16h b;
        b[0]=(_Float16)f0.x;  b[1]=(_Float16)f0.y;  b[2]=(_Float16)f0.z;  b[3]=(_Float16)f0.w;
        b[4]=(_Float16)f1.x;  b[5]=(_Float16)f1.y;  b[6]=(_Float16)f1.z;  b[7]=(_Float16)f1.w;
        b[8]=(_Float16)f2.x;  b[9]=(_Float16)f2.y;  b[10]=(_Float16)f2.z; b[11]=(_Float16)f2.w;
        b[12]=(_Float16)f3.x; b[13]=(_Float16)f3.y; b[14]=(_Float16)f3.z; b[15]=(_Float16)f3.w;
        AFrag a0, a1;
        const _Float16* A0 = &As[l15 * LDSP + k0 + khalf * 8];
        a0.q[0] = *(const uint4*)(A0);
        a0.q[1] = *(const uint4*)(A0 + 16);
        const _Float16* A1 = &As[(16 + l15) * LDSP + k0 + khalf * 8];
        a1.q[0] = *(const uint4*)(A1);
        a1.q[1] = *(const uint4*)(A1 + 16);
        c0 = __builtin_amdgcn_wmma_f32_16x16x32_f16(false, a0.h, false, b, (short)0, c0, false, false);
        c1 = __builtin_amdgcn_wmma_f32_16x16x32_f16(false, a1.h, false, b, (short)0, c1, false, false);
      }
    }

    // epilogue for this 16-wide tile: shifted exp + positive mask
    #pragma unroll
    for (int v = 0; v < 8; ++v) {
      float lg0 = c0[v] * INV_T;
      eacc[v] += __expf(lg0 - SHIFT);
      if (ml == lab[v])     { pacc[v]     += lg0; cacc[v]     += 1.0f; }
      float lg1 = c1[v] * INV_T;
      eacc[8 + v] += __expf(lg1 - SHIFT);
      if (ml == lab[8 + v]) { pacc[8 + v] += lg1; cacc[8 + v] += 1.0f; }
    }
  }

  // butterfly reduce across the 16-lane halves (C columns live across lanes)
  #pragma unroll
  for (int i = 0; i < 16; ++i) {
    #pragma unroll
    for (int m = 8; m >= 1; m >>= 1) {
      eacc[i] += __shfl_xor(eacc[i], m, 32);
      pacc[i] += __shfl_xor(pacc[i], m, 32);
      cacc[i] += __shfl_xor(cacc[i], m, 32);
    }
  }

  if (l15 == 0) {
    const int slotcol = nchunk * 8 + wave;      // 0..63
    #pragma unroll
    for (int i = 0; i < 16; ++i) {
      int s = i >> 3, v = i & 7;
      int anchor = mb * MBLK + s * 16 + khalf * 8 + v;
      size_t base = ((size_t)(p * BATCH + anchor)) * 64 + slotcol;
      slots[base]          = eacc[i];
      slots[65536 + base]  = pacc[i];
      slots[131072 + base] = cacc[i];
    }
  }
}

// ---------------------------------------------------------------------------
// Kernel 3: deterministic fixed-order combine -> 2 output losses.
// 1 block, 256 threads (thread t = anchor t)
// ---------------------------------------------------------------------------
__global__ void supcon_finalize(const float* __restrict__ slots,
                                float* __restrict__ out) {
  const int t = threadIdx.x;
  float lossP[4];
  #pragma unroll
  for (int p = 0; p < 4; ++p) {
    size_t base = ((size_t)(p * BATCH + t)) * 64;
    float se = 0.f, ps = 0.f, cs = 0.f;
    for (int j = 0; j < 64; ++j) {
      se += slots[base + j];
      ps += slots[65536 + base + j];
      cs += slots[131072 + base + j];
    }
    float lse  = SHIFT + __logf(se);                 // raw logsumexp
    float mlpp = (ps - cs * lse) / fmaxf(cs, 1.0f);  // mean log-prob of positives
    lossP[p] = -mlpp * (1.0f / (float)BATCH);
  }
  __shared__ float red[256];
  __shared__ float res[4];
  for (int p = 0; p < 4; ++p) {
    red[t] = lossP[p];
    __syncthreads();
    for (int off = 128; off >= 1; off >>= 1) {
      if (t < off) red[t] += red[t + off];
      __syncthreads();
    }
    if (t == 0) res[p] = red[0];
    __syncthreads();
  }
  if (t == 0) {
    out[0] = res[0] + res[1];   // loss_contr       (rgb-rgb + ir-ir)
    out[1] = res[2] + res[3];   // loss_contr_cross (rgb-ir  + ir-rgb)
  }
}

// ---------------------------------------------------------------------------
extern "C" void kernel_launch(void* const* d_in, const int* in_sizes, int n_in,
                              void* d_out, int out_size, void* d_ws, size_t ws_size,
                              hipStream_t stream) {
  const float* in_rgb   = (const float*)d_in[0];
  const float* in_ir    = (const float*)d_in[1];
  const int*   tgt_rgb  = (const int*)  d_in[2];
  const int*   tgt_ir   = (const int*)  d_in[3];
  const float* feat_rgb = (const float*)d_in[4];
  const float* feat_ir  = (const float*)d_in[5];
  const int*   pl_rgb   = (const int*)  d_in[6];
  const int*   pl_ir    = (const int*)  d_in[7];

  float*    slots = (float*)d_ws;
  _Float16* featH = (_Float16*)((char*)d_ws + WS_FEATH);
  _Float16* memH  = (_Float16*)((char*)d_ws + WS_MEMH);

  supcon_prep<<<dim3(512), dim3(256), 0, stream>>>(in_rgb, in_ir, featH);

  dim3 grid(8, 8, 4);   // n-chunks, m-blocks, pairings
  if (ws_size >= (size_t)WS_NEED) {
    supcon_convbank<<<dim3(3072, 2), dim3(256), 0, stream>>>(feat_rgb, feat_ir, memH);
    supcon_gemm<true><<<grid, dim3(256), 0, stream>>>(featH, memH, feat_rgb, feat_ir,
                                                      tgt_rgb, tgt_ir, pl_rgb, pl_ir, slots);
  } else {
    supcon_gemm<false><<<grid, dim3(256), 0, stream>>>(featH, memH, feat_rgb, feat_ir,
                                                       tgt_rgb, tgt_ir, pl_rgb, pl_ir, slots);
  }

  supcon_finalize<<<dim3(1), dim3(256), 0, stream>>>(slots, (float*)d_out);
}
